// RNNCellMTSNet_33380485825229
// MI455X (gfx1250) — matrix operand order repaired and checked by reading, not compiled, not run on hardware
//
#include <hip/hip_runtime.h>
#include <math.h>

// ---------------- problem constants ----------------
#define T_STEPS 256
#define BATCH   128
#define IN_DIM  1024
#define HID     1024
#define NCLS    16
#define N_WGS   16          // workgroups in persistent recurrent kernel

typedef __attribute__((ext_vector_type(16))) __bf16 v16bf;
typedef __attribute__((ext_vector_type(8)))  __bf16 v8bf;
typedef __attribute__((ext_vector_type(4)))  __bf16 v4bf;
typedef __attribute__((ext_vector_type(8)))  float  v8f;
typedef __attribute__((ext_vector_type(4)))  float  v4f;

union FragB16 { v16bf v; v8bf h[2]; };

// =====================================================================
// Kernel 1: xw[t*B+b, h] = x[t,b,:] . W_ih[h,:] + b_ih[h] + b_hh[h]
// M=32768, N=1024, K=1024.  128x128 block tile, BK=32, 8 waves (4Mx2N),
// each wave does 2x4 v_wmma_f32_16x16x32_bf16 tiles.
// Grid: blockIdx.x = N (fast) so the 8 N-blocks sharing one x M-tile are
// adjacent in dispatch order -> x read from HBM ~once (L2 reuse).
// LDS tiles are double-buffered: one barrier per K-step, global loads for
// k+1 and the LDS write of k+1 overlap WMMA of k.
// =====================================================================
#define LDS_STRIDE 40   // bf16 per LDS row (32 data + 8 pad) -> 20 dwords, gcd(20,64)=4 -> conflict-free

__global__ __launch_bounds__(256) void gemm_xw_kernel(
    const float* __restrict__ x, const float* __restrict__ W_ih,
    const float* __restrict__ b_ih, const float* __restrict__ b_hh,
    float* __restrict__ xw)
{
    __shared__ __bf16 sA[2][128 * LDS_STRIDE];
    __shared__ __bf16 sB[2][128 * LDS_STRIDE];

    const int tid   = threadIdx.x;
    const int lane  = tid & 31;
    const int wave  = tid >> 5;
    const int waveM = wave >> 1;          // 0..3 -> 32 rows each
    const int waveN = wave & 1;           // 0..1 -> 64 cols each
    const int lrow  = lane & 15;
    const int hi    = lane >> 4;          // lane half

    const int blockN = blockIdx.x * 128;  // over H   (fast-varying)
    const int blockM = blockIdx.y * 128;  // over T*B (slow-varying)

    // staging: thread -> (row, 16-col half) of the 128x32 fp32 tile
    const int srow  = tid >> 1;
    const int shalf = (tid & 1) * 16;

    v8f acc[2][4];
#pragma unroll
    for (int i = 0; i < 2; ++i)
#pragma unroll
        for (int j = 0; j < 4; ++j) acc[i][j] = (v8f)0.f;

    const float* xrow = x    + (size_t)(blockM + srow) * IN_DIM + shalf;
    const float* wrow = W_ih + (size_t)(blockN + srow) * IN_DIM + shalf;

    // ---- prologue: stage k0 = 0 into buffer 0 ----
    v4f ax[4], bw[4];
#pragma unroll
    for (int i = 0; i < 4; ++i) ax[i] = *(const v4f*)(xrow + i * 4);
#pragma unroll
    for (int i = 0; i < 4; ++i) bw[i] = *(const v4f*)(wrow + i * 4);
    {
        v8bf pa0, pa1, pb0, pb1;
#pragma unroll
        for (int j = 0; j < 4; ++j) {
            pa0[j] = (__bf16)ax[0][j]; pa0[4 + j] = (__bf16)ax[1][j];
            pa1[j] = (__bf16)ax[2][j]; pa1[4 + j] = (__bf16)ax[3][j];
            pb0[j] = (__bf16)bw[0][j]; pb0[4 + j] = (__bf16)bw[1][j];
            pb1[j] = (__bf16)bw[2][j]; pb1[4 + j] = (__bf16)bw[3][j];
        }
        *(v8bf*)&sA[0][srow * LDS_STRIDE + shalf    ] = pa0;
        *(v8bf*)&sA[0][srow * LDS_STRIDE + shalf + 8] = pa1;
        *(v8bf*)&sB[0][srow * LDS_STRIDE + shalf    ] = pb0;
        *(v8bf*)&sB[0][srow * LDS_STRIDE + shalf + 8] = pb1;
    }

    int cur = 0;
    for (int k0 = 0; k0 < IN_DIM; k0 += 32) {
        const bool more = (k0 + 32) < IN_DIM;
        // ---- issue global loads for next K-step (hide behind WMMA) ----
        if (more) {
#pragma unroll
            for (int i = 0; i < 4; ++i) ax[i] = *(const v4f*)(xrow + k0 + 32 + i * 4);
#pragma unroll
            for (int i = 0; i < 4; ++i) bw[i] = *(const v4f*)(wrow + k0 + 32 + i * 4);
        }

        __syncthreads();   // buffer `cur` fully written; prior reads of `nxt` done

        // ---- fragments + WMMA from buffer `cur` ----
        FragB16 a[2], b[4];
#pragma unroll
        for (int at = 0; at < 2; ++at) {
            const __bf16* base = &sA[cur][(waveM * 32 + at * 16 + lrow) * LDS_STRIDE];
            a[at].h[0] = *(const v8bf*)(base + hi * 8);        // K = hi*8 .. +7
            a[at].h[1] = *(const v8bf*)(base + 16 + hi * 8);   // K = 16+hi*8 ..
        }
#pragma unroll
        for (int bt = 0; bt < 4; ++bt) {
            const __bf16* base = &sB[cur][(waveN * 64 + bt * 16 + lrow) * LDS_STRIDE];
            b[bt].h[0] = *(const v8bf*)(base + hi * 8);        // K = hi*8 .. +7
            b[bt].h[1] = *(const v8bf*)(base + 16 + hi * 8);   // K = 16+hi*8 ..
        }
#pragma unroll
        for (int at = 0; at < 2; ++at)
#pragma unroll
            for (int bt = 0; bt < 4; ++bt)
                acc[at][bt] = __builtin_amdgcn_wmma_f32_16x16x32_bf16(
                    false, a[at].v, false, b[bt].v,
                    (short)0, acc[at][bt], false, false);

        // ---- convert & stage next K-step into buffer `cur^1` ----
        if (more) {
            const int nxt = cur ^ 1;
            v8bf pa0, pa1, pb0, pb1;
#pragma unroll
            for (int j = 0; j < 4; ++j) {
                pa0[j] = (__bf16)ax[0][j]; pa0[4 + j] = (__bf16)ax[1][j];
                pa1[j] = (__bf16)ax[2][j]; pa1[4 + j] = (__bf16)ax[3][j];
                pb0[j] = (__bf16)bw[0][j]; pb0[4 + j] = (__bf16)bw[1][j];
                pb1[j] = (__bf16)bw[2][j]; pb1[4 + j] = (__bf16)bw[3][j];
            }
            *(v8bf*)&sA[nxt][srow * LDS_STRIDE + shalf    ] = pa0;
            *(v8bf*)&sA[nxt][srow * LDS_STRIDE + shalf + 8] = pa1;
            *(v8bf*)&sB[nxt][srow * LDS_STRIDE + shalf    ] = pb0;
            *(v8bf*)&sB[nxt][srow * LDS_STRIDE + shalf + 8] = pb1;
        }
        cur ^= 1;
    }

    // ---- epilogue: add biases, store fp32 xw ----
#pragma unroll
    for (int bt = 0; bt < 4; ++bt) {
        const int col  = blockN + waveN * 64 + bt * 16 + lrow;
        const float bias = b_ih[col] + b_hh[col];
#pragma unroll
        for (int at = 0; at < 2; ++at) {
            const int row = blockM + waveM * 32 + at * 16 + hi * 8;
            float* outp = xw + (size_t)row * HID + col;
#pragma unroll
            for (int r = 0; r < 8; ++r)
                outp[(size_t)r * HID] = acc[at][bt][r] + bias;
        }
    }
}

// =====================================================================
// Kernel 2: persistent recurrence.  16 WGs, WG g owns columns [g*64, g*64+64).
// W_hh slice lives in LDS (bf16, padded) for all 256 steps.  h double-buffered
// bf16 in global (fits L2).  Grid-wide step barrier via monotonic counter.
// =====================================================================
#define WHH_STRIDE 1032   // bf16/row: 1024 + 8 pad -> 516 dwords, 4n mod 64 banks

__global__ __launch_bounds__(256) void rnn_steps_kernel(
    const float* __restrict__ xw, const float* __restrict__ W_hh,
    __bf16* hbuf,                     // 2 * BATCH*HID bf16 (ping-pong)
    float* __restrict__ h_f32,        // final-step fp32 h
    unsigned int* ctr)
{
    extern __shared__ __bf16 sW[];    // 64 * WHH_STRIDE bf16 = 129 KB

    const int tid   = threadIdx.x;
    const int lane  = tid & 31;
    const int wave  = tid >> 5;
    const int waveM = wave >> 1;      // 0..3 -> 32 rows
    const int waveN = wave & 1;       // 0..1 -> 32 cols
    const int lrow  = lane & 15;
    const int hi    = lane >> 4;
    const int nbase = blockIdx.x * 64;

    // ---- load this WG's W_hh slice into LDS as bf16 (once) ----
    for (int n = 0; n < 64; ++n) {
        v4f w = *(const v4f*)(W_hh + (size_t)(nbase + n) * HID + tid * 4);
        v4bf p;
#pragma unroll
        for (int j = 0; j < 4; ++j) p[j] = (__bf16)w[j];
        *(v4bf*)&sW[n * WHH_STRIDE + tid * 4] = p;
    }
    __syncthreads();

    for (int t = 0; t < T_STEPS; ++t) {
        const __bf16* h_read  = hbuf + (size_t)(t & 1) * BATCH * HID;
        __bf16*       h_write = hbuf + (size_t)((t + 1) & 1) * BATCH * HID;
        const float*  xwt     = xw + (size_t)t * BATCH * HID;

        v8f acc[2][2];
#pragma unroll
        for (int i = 0; i < 2; ++i)
#pragma unroll
            for (int j = 0; j < 2; ++j) acc[i][j] = (v8f)0.f;

        for (int k0 = 0; k0 < HID; k0 += 32) {
            FragB16 a[2], b[2];
#pragma unroll
            for (int at = 0; at < 2; ++at) {
                const __bf16* ap =
                    h_read + (size_t)(waveM * 32 + at * 16 + lrow) * HID + k0;
                a[at].h[0] = *(const v8bf*)(ap + hi * 8);
                a[at].h[1] = *(const v8bf*)(ap + 16 + hi * 8);
            }
#pragma unroll
            for (int bt = 0; bt < 2; ++bt) {
                const __bf16* bp =
                    &sW[(waveN * 32 + bt * 16 + lrow) * WHH_STRIDE + k0];
                b[bt].h[0] = *(const v8bf*)(bp + hi * 8);
                b[bt].h[1] = *(const v8bf*)(bp + 16 + hi * 8);
            }
#pragma unroll
            for (int at = 0; at < 2; ++at)
#pragma unroll
                for (int bt = 0; bt < 2; ++bt)
                    acc[at][bt] = __builtin_amdgcn_wmma_f32_16x16x32_bf16(
                        false, a[at].v, false, b[bt].v,
                        (short)0, acc[at][bt], false, false);
        }

        // prefetch next step's xw slice into cache (global_prefetch_b8)
        if (t + 1 < T_STEPS)
            __builtin_prefetch(xwt + (size_t)BATCH * HID +
                               (size_t)(waveM * 32 + lrow) * HID + nbase, 0, 1);

        // ---- epilogue: add xw_t, tanh, write bf16 h (and fp32 on last step) ----
#pragma unroll
        for (int bt = 0; bt < 2; ++bt) {
            const int col = nbase + waveN * 32 + bt * 16 + lrow;
#pragma unroll
            for (int at = 0; at < 2; ++at) {
                const int row = waveM * 32 + at * 16 + hi * 8;
#pragma unroll
                for (int r = 0; r < 8; ++r) {
                    float v  = acc[at][bt][r] + xwt[(size_t)(row + r) * HID + col];
                    float hv = tanhf(v);
                    h_write[(size_t)(row + r) * HID + col] = (__bf16)hv;
                    if (t == T_STEPS - 1)
                        h_f32[(size_t)(row + r) * HID + col] = hv;
                }
            }
        }

        // ---- grid-wide step barrier (monotonic counter) ----
        __threadfence();                 // release all h stores to device scope
        __syncthreads();
        if (tid == 0) {
            __hip_atomic_fetch_add(ctr, 1u, __ATOMIC_RELEASE,
                                   __HIP_MEMORY_SCOPE_AGENT);
            const unsigned target = (unsigned)(N_WGS * (t + 1));
            while (__hip_atomic_load(ctr, __ATOMIC_ACQUIRE,
                                     __HIP_MEMORY_SCOPE_AGENT) < target)
                __builtin_amdgcn_s_sleep(2);
        }
        __syncthreads();
        __threadfence();                 // acquire: invalidate stale h lines
    }
}

// =====================================================================
// Kernel 3: out[b,c] = sigmoid(h[b,:] . W_fc[c,:] + b_fc[c])
// =====================================================================
__global__ __launch_bounds__(128) void fc_sigmoid_kernel(
    const float* __restrict__ h, const float* __restrict__ W_fc,
    const float* __restrict__ b_fc, float* __restrict__ out)
{
    __shared__ float wrow[HID];
    const int c   = blockIdx.x;
    const int tid = threadIdx.x;
    for (int i = tid * 4; i < HID; i += 128 * 4)
        *(v4f*)&wrow[i] = *(const v4f*)&W_fc[(size_t)c * HID + i];
    __syncthreads();

    float sum = 0.f;
    const float* hr = h + (size_t)tid * HID;
    for (int k = 0; k < HID; k += 4) {
        v4f hv = *(const v4f*)(hr + k);
        v4f wv = *(const v4f*)&wrow[k];
        sum += hv[0] * wv[0] + hv[1] * wv[1] + hv[2] * wv[2] + hv[3] * wv[3];
    }
    sum += b_fc[c];
    out[(size_t)tid * NCLS + c] = 1.f / (1.f + __expf(-sum));
}

// =====================================================================
extern "C" void kernel_launch(void* const* d_in, const int* in_sizes, int n_in,
                              void* d_out, int out_size, void* d_ws, size_t ws_size,
                              hipStream_t stream)
{
    const float* x    = (const float*)d_in[0];
    const float* W_ih = (const float*)d_in[1];
    const float* W_hh = (const float*)d_in[2];
    const float* b_ih = (const float*)d_in[3];
    const float* b_hh = (const float*)d_in[4];
    const float* W_fc = (const float*)d_in[5];
    const float* b_fc = (const float*)d_in[6];
    float* out = (float*)d_out;

    char* ws = (char*)d_ws;
    float* xw = (float*)ws;                                       // 128 MB fp32
    size_t off = (size_t)T_STEPS * BATCH * HID * sizeof(float);
    __bf16* hbuf = (__bf16*)(ws + off);                           // 2x ping-pong h
    off += 2 * (size_t)BATCH * HID * sizeof(__bf16);
    float* h_f32 = (float*)(ws + off);
    off += (size_t)BATCH * HID * sizeof(float);
    unsigned* ctr = (unsigned*)(ws + off);

    // h(0) = 0 in buffer 0; barrier counter = 0 (graph-capture safe memsets)
    hipMemsetAsync(hbuf, 0, (size_t)BATCH * HID * sizeof(__bf16), stream);
    hipMemsetAsync(ctr, 0, sizeof(unsigned), stream);

    // N fast-varying, M slow-varying: consecutive blocks share the x M-tile
    dim3 g1(HID / 128, (T_STEPS * BATCH) / 128);                  // 8 x 256
    gemm_xw_kernel<<<g1, 256, 0, stream>>>(x, W_ih, b_ih, b_hh, xw);

    const size_t smem = (size_t)64 * WHH_STRIDE * sizeof(__bf16); // ~129 KB of 320 KB LDS
    rnn_steps_kernel<<<N_WGS, 256, smem, stream>>>(xw, W_hh, hbuf, h_f32, ctr);

    fc_sigmoid_kernel<<<NCLS, 128, 0, stream>>>(h_f32, W_fc, b_fc, out);
}
